// LSTMDecoder_57543971832694
// MI455X (gfx1250) — matrix operand order repaired and checked by reading, not compiled, not run on hardware
//
#include <hip/hip_runtime.h>

#define BB   256
#define LATD 128
#define HD   256
#define OUTD 64
#define SS   512
#define G4H  1024   // 4*H

typedef __attribute__((ext_vector_type(16))) _Float16 v16h;
typedef __attribute__((ext_vector_type(8)))  _Float16 v8h;
typedef __attribute__((ext_vector_type(8)))  float    v8f;

// ---------------- WMMA tile loaders (CDNA5 §7.12.2 layouts) ----------------

// A: 16x32 f16 tile from row-major [m, k] memory (ld in halves).
// lane l: M = l&15 ; elems 0..7 -> K = (l>>4)*8 + e ; elems 8..15 -> K = 16 + (l>>4)*8 + (e-8)
__device__ inline v16h load_a_tile(const _Float16* base, int ld, int lane) {
    const int m  = lane & 15;
    const int kb = (lane >> 4) * 8;
    const _Float16* p = base + (size_t)m * ld + kb;
    v8h lo = *(const v8h*)(p);
    v8h hi = *(const v8h*)(p + 16);
    v16h r;
#pragma unroll
    for (int i = 0; i < 8; ++i) { r[i] = lo[i]; r[i + 8] = hi[i]; }
    return r;
}

// B: 32x16 f16 tile, sourced from weight matrix stored [n, k] row-major
// (B^T row-major == B column-major, which is what WMMA wants).
// lane l: N = l&15 ; elems 0..7 -> K = (l>>4)*8 + e ; elems 8..15 -> K = 16 + (l>>4)*8 + (e-8)
__device__ inline v16h load_b_tile(const _Float16* base, int ld, int lane) {
    const int n  = lane & 15;
    const int kb = (lane >> 4) * 8;
    const _Float16* p = base + (size_t)n * ld + kb;
    v8h lo = *(const v8h*)(p);
    v8h hi = *(const v8h*)(p + 16);
    v16h r;
#pragma unroll
    for (int i = 0; i < 8; ++i) { r[i] = lo[i]; r[i + 8] = hi[i]; }
    return r;
}

__device__ inline float fast_sigmoid(float x) { return 1.0f / (1.0f + __expf(-x)); }
__device__ inline float fast_tanh(float x) {
    float t = __expf(-2.0f * x);
    return (1.0f - t) / (1.0f + t);
}

// ---------------- init / small kernels ----------------

__global__ void cvt_f32_to_f16_kernel(const float* __restrict__ in, _Float16* __restrict__ out, int n) {
    int i = blockIdx.x * blockDim.x + threadIdx.x;
    if (i < n) out[i] = (_Float16)in[i];
}

// hidden/cell initial states: [B, 2H] each; split into layer 0/1 slabs.
__global__ void init_states_kernel(const float* __restrict__ z,
                                   const float* __restrict__ hw, const float* __restrict__ hb,
                                   const float* __restrict__ cw, const float* __restrict__ cb,
                                   _Float16* __restrict__ h0h /*[2*B*H] f16*/,
                                   float* __restrict__ c0 /*[2*B*H] f32*/) {
    int i = blockIdx.x * blockDim.x + threadIdx.x;
    if (i >= BB * 2 * HD) return;
    int b = i / (2 * HD), j = i % (2 * HD);
    int layer = j >> 8, hh = j & (HD - 1);
    float ah = hb[j], ac = cb[j];
    for (int k = 0; k < LATD; ++k) {
        float zv = z[b * LATD + k];
        ah += zv * hw[(size_t)j * LATD + k];
        ac += zv * cw[(size_t)j * LATD + k];
    }
    h0h[(size_t)layer * BB * HD + (size_t)b * HD + hh] = (_Float16)ah;
    c0 [(size_t)layer * BB * HD + (size_t)b * HD + hh] = ac;
}

__global__ void x0_kernel(const float* __restrict__ z, const float* __restrict__ iw,
                          const float* __restrict__ ib, float* __restrict__ x0) {
    int i = blockIdx.x * blockDim.x + threadIdx.x;
    if (i >= BB * HD) return;
    int b = i / HD, hh = i % HD;
    float a = ib[hh];
    for (int k = 0; k < LATD; ++k) a += z[b * LATD + k] * iw[(size_t)hh * LATD + k];
    x0[i] = a;
}

// gx0[b,n] = x0[b,:]·W_ih0[n,:] + b_ih0[n] + b_hh0[n]  (constant over all timesteps)
__global__ void gx0_kernel(const float* __restrict__ x0, const float* __restrict__ Wih0,
                           const float* __restrict__ bih0, const float* __restrict__ bhh0,
                           float* __restrict__ gx0) {
    int i = blockIdx.x * blockDim.x + threadIdx.x;
    if (i >= BB * G4H) return;
    int b = i / G4H, n = i % G4H;
    float a = bih0[n] + bhh0[n];
    for (int k = 0; k < HD; ++k) a += x0[b * HD + k] * Wih0[(size_t)n * HD + k];
    gx0[i] = a;
}

// ---------------- fused 2-layer LSTM recurrence ----------------
// One workgroup per 16-row batch tile (recurrence is batch-independent -> only
// workgroup barriers, no grid sync). 16 waves; wave w owns h-columns [16w,16w+16)
// for both layers. Hidden states live in LDS (double buffered); cell states and
// the time-invariant layer-0 gate inputs live in registers for all 512 steps.
// NOTE: weight pointers intentionally NOT __restrict__ and a per-step memory
// clobber is used so the compiler cannot hoist the (loop-invariant) weight tile
// loads out of the 512-step loop — hoisting them overflows the register file
// and turns into scratch spill/reload in the critical sequential chain. The
// weights are L2-resident; re-issuing global_load_b128 each step is the fast path.

__global__ __launch_bounds__(512) void lstm_fused_kernel(
    const _Float16* __restrict__ h0l0,   // [B*H] layer0 initial hidden (f16)
    const _Float16* __restrict__ h0l1,   // [B*H] layer1 initial hidden (f16)
    const float*    __restrict__ c0l0,   // [B*H] layer0 initial cell
    const float*    __restrict__ c0l1,   // [B*H] layer1 initial cell
    const _Float16* Whh0,                // [4H*H] f16 (no restrict: see note)
    const _Float16* Wih1,                // [4H*H] f16
    const _Float16* Whh1,                // [4H*H] f16
    const float*    __restrict__ gx0,    // [B*4H] f32 (layer0 gate input, time-invariant)
    const float*    __restrict__ bih1, const float* __restrict__ bhh1,
    _Float16*       __restrict__ h2seq)  // out: [S*B*H] f16 (row = s*B + b)
{
    __shared__ _Float16 sh_h1[2][16 * HD];   // layer-0 hidden, double buffered (8KB x2)
    __shared__ _Float16 sh_h2[2][16 * HD];   // layer-1 hidden, double buffered (8KB x2)

    const int tid   = threadIdx.x;
    const int lane  = tid & 31;
    const int w     = tid >> 5;              // wave 0..15 -> h-column tile
    const int b0    = blockIdx.x * 16;       // batch rows of this block
    const int hc0   = w * 16;
    const int lhalf = lane >> 4;             // C/D: m = r + 8*lhalf
    const int cn    = lane & 15;             // C/D: n = cn

    // seed LDS hidden buffers (parity 0) with initial states
    for (int i = tid; i < 16 * HD; i += 512) {
        sh_h1[0][i] = h0l0[(size_t)b0 * HD + i];
        sh_h2[0][i] = h0l1[(size_t)b0 * HD + i];
    }

    // persistent per-wave register state
    float c0reg[8], c1reg[8];
#pragma unroll
    for (int r = 0; r < 8; ++r) {
        size_t idx = (size_t)(b0 + r + 8 * lhalf) * HD + hc0 + cn;
        c0reg[r] = c0l0[idx];
        c1reg[r] = c0l1[idx];
    }
    v8f gx0hold[4];                          // layer-0 gate inputs for this tile
#pragma unroll
    for (int g = 0; g < 4; ++g)
#pragma unroll
        for (int r = 0; r < 8; ++r)
            gx0hold[g][r] = gx0[(size_t)(b0 + r + 8 * lhalf) * G4H + g * HD + hc0 + cn];
    float bias1[4];
#pragma unroll
    for (int g = 0; g < 4; ++g)
        bias1[g] = bih1[g * HD + hc0 + cn] + bhh1[g * HD + hc0 + cn];

    __syncthreads();

    for (int s = 0; s < SS; ++s) {
        // Stop LICM of the weight-tile loads (would spill to scratch).
        asm volatile("" ::: "memory");

        const int pr = s & 1;                // read parity
        const int cu = pr ^ 1;               // write parity

        // ---- phase A: layer 0 -> h1[cu] ----
        {
            v8f acc[4];
#pragma unroll
            for (int g = 0; g < 4; ++g) acc[g] = gx0hold[g];
#pragma unroll
            for (int kc = 0; kc < 8; ++kc) {
                v16h a = load_a_tile(&sh_h1[pr][kc * 32], HD, lane);
#pragma unroll
                for (int g = 0; g < 4; ++g) {
                    v16h bm = load_b_tile(Whh0 + (size_t)(g * HD + hc0) * HD + kc * 32, HD, lane);
                    acc[g] = __builtin_amdgcn_wmma_f32_16x16x32_f16(false, a, false, bm,
                                                                    (short)0, acc[g], false, false);
                }
            }
#pragma unroll
            for (int r = 0; r < 8; ++r) {
                float ig = fast_sigmoid(acc[0][r]);
                float fg = fast_sigmoid(acc[1][r]);
                float gg = fast_tanh(acc[2][r]);
                float og = fast_sigmoid(acc[3][r]);
                c0reg[r] = fg * c0reg[r] + ig * gg;
                float hv = og * fast_tanh(c0reg[r]);
                sh_h1[cu][(r + 8 * lhalf) * HD + hc0 + cn] = (_Float16)hv;
            }
        }
        __syncthreads();

        // ---- phase B: layer 1 (fused input + recurrent GEMMs) -> h2[cu] ----
        {
            v8f acc[4];
#pragma unroll
            for (int g = 0; g < 4; ++g)
#pragma unroll
                for (int r = 0; r < 8; ++r) acc[g][r] = bias1[g];
#pragma unroll
            for (int kc = 0; kc < 8; ++kc) {
                v16h a1 = load_a_tile(&sh_h1[cu][kc * 32], HD, lane);   // layer-0 output, this step
                v16h a2 = load_a_tile(&sh_h2[pr][kc * 32], HD, lane);   // layer-1 hidden, prev step
#pragma unroll
                for (int g = 0; g < 4; ++g) {
                    v16h bi = load_b_tile(Wih1 + (size_t)(g * HD + hc0) * HD + kc * 32, HD, lane);
                    acc[g] = __builtin_amdgcn_wmma_f32_16x16x32_f16(false, a1, false, bi,
                                                                    (short)0, acc[g], false, false);
                    v16h bh = load_b_tile(Whh1 + (size_t)(g * HD + hc0) * HD + kc * 32, HD, lane);
                    acc[g] = __builtin_amdgcn_wmma_f32_16x16x32_f16(false, a2, false, bh,
                                                                    (short)0, acc[g], false, false);
                }
            }
            _Float16* hout = h2seq + (size_t)s * BB * HD;
#pragma unroll
            for (int r = 0; r < 8; ++r) {
                float ig = fast_sigmoid(acc[0][r]);
                float fg = fast_sigmoid(acc[1][r]);
                float gg = fast_tanh(acc[2][r]);
                float og = fast_sigmoid(acc[3][r]);
                c1reg[r] = fg * c1reg[r] + ig * gg;
                float hv = og * fast_tanh(c1reg[r]);
                _Float16 hf = (_Float16)hv;
                sh_h2[cu][(r + 8 * lhalf) * HD + hc0 + cn] = hf;
                hout[(size_t)(b0 + r + 8 * lhalf) * HD + hc0 + cn] = hf;
            }
        }
        __syncthreads();
    }
}

// ---------------- output projection: [(S*B) x H] f16 x Wo^T -> fp32 [B,S,OUT] ----------------

__global__ __launch_bounds__(256) void out_gemm_kernel(
    const _Float16* __restrict__ h2seq, // row = s*B + b, K = H contiguous
    const _Float16* __restrict__ Wo,    // [OUT x H] f16
    const float*    __restrict__ bo,    // [OUT]
    float*          __restrict__ out)   // [B, S, OUT]
{
    const int lane  = threadIdx.x & 31;
    const int wid   = threadIdx.x >> 5;
    const int gw    = blockIdx.x * 8 + wid;   // 0..32767 tiles
    const int row0  = (gw >> 2) * 16;
    const int n0    = (gw & 3) * 16;
    const int lhalf = lane >> 4;
    const int cn    = lane & 15;

    v8f acc;
#pragma unroll
    for (int r = 0; r < 8; ++r) acc[r] = bo[n0 + cn];

#pragma unroll
    for (int kc = 0; kc < 8; ++kc) {
        v16h a = load_a_tile(h2seq + (size_t)row0 * HD + kc * 32, HD, lane);
        v16h b = load_b_tile(Wo + (size_t)n0 * HD + kc * 32, HD, lane);
        acc = __builtin_amdgcn_wmma_f32_16x16x32_f16(false, a, false, b, (short)0, acc, false, false);
    }

#pragma unroll
    for (int r = 0; r < 8; ++r) {
        int m = row0 + r + 8 * lhalf;       // m = s*B + b
        int srow = m >> 8;                  // / 256
        int brow = m & 255;
        out[(size_t)brow * (SS * OUTD) + (size_t)srow * OUTD + n0 + cn] = acc[r];
    }
}

// ---------------- host launcher ----------------

extern "C" void kernel_launch(void* const* d_in, const int* in_sizes, int n_in,
                              void* d_out, int out_size, void* d_ws, size_t ws_size,
                              hipStream_t stream) {
    const float* z      = (const float*)d_in[0];
    const float* fhw    = (const float*)d_in[1];
    const float* fhb    = (const float*)d_in[2];
    const float* fcw    = (const float*)d_in[3];
    const float* fcb    = (const float*)d_in[4];
    const float* fiw    = (const float*)d_in[5];
    const float* fib    = (const float*)d_in[6];
    const float* Wih0   = (const float*)d_in[7];
    const float* Whh0   = (const float*)d_in[8];
    const float* bih0   = (const float*)d_in[9];
    const float* bhh0   = (const float*)d_in[10];
    const float* Wih1   = (const float*)d_in[11];
    const float* Whh1   = (const float*)d_in[12];
    const float* bih1   = (const float*)d_in[13];
    const float* bhh1   = (const float*)d_in[14];
    const float* Wo     = (const float*)d_in[15];
    const float* bo     = (const float*)d_in[16];
    float* out          = (float*)d_out;

    // workspace carve-up (256B aligned slabs)
    char* ws = (char*)d_ws;
    size_t off = 0;
    auto take = [&](size_t bytes) -> void* {
        void* p = ws + off;
        off = (off + bytes + 255) & ~(size_t)255;
        return p;
    };
    float*     x0    = (float*)take((size_t)BB * HD * 4);
    float*     gx0   = (float*)take((size_t)BB * G4H * 4);
    _Float16*  h0h   = (_Float16*)take((size_t)2 * BB * HD * 2);
    float*     c0    = (float*)take((size_t)2 * BB * HD * 4);
    _Float16*  whh0h = (_Float16*)take((size_t)G4H * HD * 2);
    _Float16*  wih1h = (_Float16*)take((size_t)G4H * HD * 2);
    _Float16*  whh1h = (_Float16*)take((size_t)G4H * HD * 2);
    _Float16*  woh   = (_Float16*)take((size_t)OUTD * HD * 2);
    _Float16*  h2seq = (_Float16*)take((size_t)SS * BB * HD * 2);
    (void)ws_size; (void)n_in; (void)in_sizes; (void)out_size;

    // 1) weights -> f16 for WMMA (stay resident in 192MB L2 across all 512 steps)
    cvt_f32_to_f16_kernel<<<(G4H * HD + 255) / 256, 256, 0, stream>>>(Whh0, whh0h, G4H * HD);
    cvt_f32_to_f16_kernel<<<(G4H * HD + 255) / 256, 256, 0, stream>>>(Wih1, wih1h, G4H * HD);
    cvt_f32_to_f16_kernel<<<(G4H * HD + 255) / 256, 256, 0, stream>>>(Whh1, whh1h, G4H * HD);
    cvt_f32_to_f16_kernel<<<(OUTD * HD + 255) / 256, 256, 0, stream>>>(Wo, woh, OUTD * HD);

    // 2) initial states, decoder input, precomputed layer-0 gate input
    init_states_kernel<<<(BB * 2 * HD + 255) / 256, 256, 0, stream>>>(z, fhw, fhb, fcw, fcb, h0h, c0);
    x0_kernel<<<(BB * HD + 255) / 256, 256, 0, stream>>>(z, fiw, fib, x0);
    gx0_kernel<<<(BB * G4H + 255) / 256, 256, 0, stream>>>(x0, Wih0, bih0, bhh0, gx0);

    // 3) fused two-layer recurrence: one workgroup per batch tile, no grid sync
    lstm_fused_kernel<<<16, 512, 0, stream>>>(
        h0h, h0h + (size_t)BB * HD, c0, c0 + (size_t)BB * HD,
        whh0h, wih1h, whh1h, gx0, bih1, bhh1, h2seq);

    // 4) output projection: 32768 tiles / 8 waves per block
    out_gemm_kernel<<<4096, 256, 0, stream>>>(h2seq, woh, bo, out);
}
